// NoExtraEvoformerIteration_51178830299629
// MI455X (gfx1250) — matrix operand order repaired and checked by reading, not compiled, not run on hardware
//
#include <hip/hip_runtime.h>
#include <hip/hip_bf16.h>

// ---------------------------------------------------------------------------
// Evoformer iteration for gfx1250 (MI455X).
// All contractions -> v_wmma_f32_16x16x32_bf16 (wave32). 64x64 macro-tile,
// 8 waves, 2 WMMA tiles per wave per K-step. f32 global operands converted to
// bf16 during vectorized LDS staging (b128 loads + ds_store_b64); staging
// variant chosen at compile time per GEMM via template dispatch. Fused
// bias/ReLU/sigmoid epilogues, fully strided A/B/C addressing.
// Required workspace: (2*67108864 + 10*16777216) floats ~= 1.21 GB.
// ---------------------------------------------------------------------------

#define NSEQ 128
#define NRES 256
#define CM   256
#define CZ   128
#define HM   8
#define HP   4
#define KDIM 32
#define NICC 128

// Input index map: top-level dict order (msa_act, pair_act, msa_mask,
// pair_mask, params) with params flattened as a pytree (sorted keys).
enum {
  I_MSA = 0, I_PAIR, I_MMASK, I_PMASK,
  MC_GATING_B, MC_GATING_W, MC_K_W, MC_NORM_O, MC_NORM_S, MC_OUTPUT_B, MC_OUTPUT_W, MC_Q_W, MC_V_W,
  MR_FEAT, MR_GATING_B, MR_GATING_W, MR_K_W, MR_NORM_O, MR_NORM_S, MR_OUTPUT_B, MR_OUTPUT_W, MR_PN_O, MR_PN_S, MR_Q_W, MR_V_W,
  MT_B1, MT_B2, MT_NORM_O, MT_NORM_S, MT_W1, MT_W2,
  OPM_LEFT_B, OPM_LEFT_W, OPM_NORM_O, OPM_NORM_S, OPM_OUT_B, OPM_OUT_W, OPM_RIGHT_B, OPM_RIGHT_W,
  PT_B1, PT_B2, PT_NORM_O, PT_NORM_S, PT_W1, PT_W2,
  TAE_FEAT, TAE_GATING_B, TAE_GATING_W, TAE_K_W, TAE_NORM_O, TAE_NORM_S, TAE_OUTPUT_B, TAE_OUTPUT_W, TAE_Q_W, TAE_V_W,
  TAS_FEAT, TAS_GATING_B, TAS_GATING_W, TAS_K_W, TAS_NORM_O, TAS_NORM_S, TAS_OUTPUT_B, TAS_OUTPUT_W, TAS_Q_W, TAS_V_W,
  TMI_C_O, TMI_C_S, TMI_GATE_B, TMI_GATE_W, TMI_GL_B, TMI_GL_W, TMI_LN_O, TMI_LN_S, TMI_OUT_B, TMI_OUT_W, TMI_PROJ_B, TMI_PROJ_W,
  TMO_C_O, TMO_C_S, TMO_GATE_B, TMO_GATE_W, TMO_GL_B, TMO_GL_W, TMO_LN_O, TMO_LN_S, TMO_OUT_B, TMO_OUT_W, TMO_PROJ_B, TMO_PROJ_W
};

typedef __attribute__((ext_vector_type(16))) __bf16 v16bf;
typedef __attribute__((ext_vector_type(4)))  __bf16 v4bf;
typedef __attribute__((ext_vector_type(8)))  float  v8f;

#define BM 64
#define BN 64
#define BK 32
#define LDT 40   // padded LDS row stride (bf16): 8B-aligned quads, conflict-free frags

struct GemmP {
  const float* A; const float* B; float* C; const float* bias;
  int  M, N, K;         // K must be a multiple of 32 (true for every call here)
  long lda, acs;        // A addr = aoff + m*lda + k*acs
  long ldb, bcs;        // B addr = boff + k*ldb + n*bcs
  long ldc, ccs;        // C addr = coff + m*ldc + n*ccs
  int  bi;              // batch inner count: zo=z/bi, zi=z%bi
  long aOut, aIn, bOut, bIn, cOut, cIn, biasIn;
  float alpha;
  int  act;             // 0 none, 1 relu, 2 sigmoid
  int  accum;           // 0 store, 1 +=
};

// AMODE: 0 = A contiguous in K (acs==1, 4-aligned)  -> b128 + ds_store_b64
//        1 = A contiguous in M (lda==1, M%BM==0)    -> b128 along M
//        2 = arbitrary strides                      -> 4 strided loads + b64
// BMODE: 0 = B contiguous in K (ldb==1, 4-aligned)  -> b128 + ds_store_b64
//        1 = arbitrary strides                      -> 4 strided loads + b64
template<int AMODE, int BMODE>
__global__ __launch_bounds__(256) void k_gemm(GemmP p) {
  __shared__ __align__(16) __bf16 As[BM * LDT];
  __shared__ __align__(16) __bf16 Bs[BN * LDT];
  const int tid  = threadIdx.x;
  const int z    = blockIdx.z;
  const int m0   = blockIdx.y * BM;
  const int n0   = blockIdx.x * BN;
  const long zo  = z / p.bi, zi = z % p.bi;
  const long aoff = zo * p.aOut + zi * p.aIn;
  const long boff = zo * p.bOut + zi * p.bIn;
  const long coff = zo * p.cOut + zi * p.cIn;
  const long biasoff = zi * p.biasIn;
  const int lane = tid & 31;
  const int wave = tid >> 5;
  const int wm   = (wave >> 1) << 4;   // 0,16,32,48
  const int wn   = (wave & 1) << 5;    // 0,32 (two 16-wide tiles per wave)
  const int half = lane >> 4;
  const int l15  = lane & 15;
  v8f accs[2];
  accs[0] = (v8f){0.f,0.f,0.f,0.f,0.f,0.f,0.f,0.f};
  accs[1] = (v8f){0.f,0.f,0.f,0.f,0.f,0.f,0.f,0.f};

  for (int k0 = 0; k0 < p.K; k0 += BK) {
    // ---------------- stage A (BM x BK) ----------------
    if constexpr (AMODE == 0) {
#pragma unroll
      for (int it = 0; it < 2; ++it) {
        int idx = tid + (it << 8);
        int m = idx >> 3, k4 = (idx & 7) << 2;
        int gm = m0 + m;
        v4bf w;
        if (gm < p.M) {
          float4 f = *(const float4*)&p.A[aoff + (long)gm * p.lda + (k0 + k4)];
          w[0] = (__bf16)f.x; w[1] = (__bf16)f.y; w[2] = (__bf16)f.z; w[3] = (__bf16)f.w;
        } else { w[0] = w[1] = w[2] = w[3] = (__bf16)0.f; }
        *(v4bf*)&As[m * LDT + k4] = w;
      }
    } else if constexpr (AMODE == 1) {
#pragma unroll
      for (int it = 0; it < 2; ++it) {
        int idx = tid + (it << 8);
        int k = idx >> 4, m4 = (idx & 15) << 2;
        float4 f = *(const float4*)&p.A[aoff + (long)(k0 + k) * p.acs + (m0 + m4)];
        As[(m4 + 0) * LDT + k] = (__bf16)f.x;
        As[(m4 + 1) * LDT + k] = (__bf16)f.y;
        As[(m4 + 2) * LDT + k] = (__bf16)f.z;
        As[(m4 + 3) * LDT + k] = (__bf16)f.w;
      }
    } else {
#pragma unroll
      for (int it = 0; it < 2; ++it) {
        int idx = tid + (it << 8);
        int m = idx & 63, kq = (idx >> 6) << 2;
        int gm = m0 + m;
        v4bf w;
        if (gm < p.M) {
          const float* src = &p.A[aoff + (long)gm * p.lda + (long)(k0 + kq) * p.acs];
          w[0] = (__bf16)src[0];
          w[1] = (__bf16)src[p.acs];
          w[2] = (__bf16)src[2 * p.acs];
          w[3] = (__bf16)src[3 * p.acs];
        } else { w[0] = w[1] = w[2] = w[3] = (__bf16)0.f; }
        *(v4bf*)&As[m * LDT + kq] = w;
      }
    }
    // ---------------- stage B (BN x BK), stored [n][k] ----------------
    if constexpr (BMODE == 0) {
#pragma unroll
      for (int it = 0; it < 2; ++it) {
        int idx = tid + (it << 8);
        int n = idx >> 3, k4 = (idx & 7) << 2;
        int gn = n0 + n;
        v4bf w;
        if (gn < p.N) {
          float4 f = *(const float4*)&p.B[boff + (long)gn * p.bcs + (k0 + k4)];
          w[0] = (__bf16)f.x; w[1] = (__bf16)f.y; w[2] = (__bf16)f.z; w[3] = (__bf16)f.w;
        } else { w[0] = w[1] = w[2] = w[3] = (__bf16)0.f; }
        *(v4bf*)&Bs[n * LDT + k4] = w;
      }
    } else {
#pragma unroll
      for (int it = 0; it < 2; ++it) {
        int idx = tid + (it << 8);
        int n = idx & 63, kq = (idx >> 6) << 2;
        int gn = n0 + n;
        v4bf w;
        if (gn < p.N) {
          const float* src = &p.B[boff + (long)(k0 + kq) * p.ldb + (long)gn * p.bcs];
          w[0] = (__bf16)src[0];
          w[1] = (__bf16)src[p.ldb];
          w[2] = (__bf16)src[2 * p.ldb];
          w[3] = (__bf16)src[3 * p.ldb];
        } else { w[0] = w[1] = w[2] = w[3] = (__bf16)0.f; }
        *(v4bf*)&Bs[n * LDT + kq] = w;
      }
    }
    // prefetch next K-tile (global_prefetch_b8 on gfx1250)
    if (k0 + BK < p.K) {
      int gm = m0 + lane;
      if (gm < p.M) __builtin_prefetch(&p.A[aoff + (long)gm * p.lda + (long)(k0 + BK) * p.acs], 0, 1);
      int gn = n0 + (tid & 63);
      if (gn < p.N) __builtin_prefetch(&p.B[boff + (long)(k0 + BK) * p.ldb + (long)gn * p.bcs], 0, 1);
    }
    __syncthreads();

    // Fragments per ISA 16-bit A/B VGPR layouts (CDNA5 ISA 7.12.2)
    union { v16bf v; unsigned u[8]; } fa, fb0, fb1;
    const unsigned* arow = (const unsigned*)(As + (wm + l15) * LDT);
    const unsigned* b0   = (const unsigned*)(Bs + (wn + l15) * LDT);
    const unsigned* b1   = (const unsigned*)(Bs + (wn + 16 + l15) * LDT);
#pragma unroll
    for (int j = 0; j < 4; ++j) {
      fa.u[j]     = arow[j + 4 * half];       // K = 2j + 8*half
      fa.u[4 + j] = arow[8 + j + 4 * half];   // K = 16 + 2j + 8*half
    }
#pragma unroll
    for (int j = 0; j < 8; ++j) { fb0.u[j] = b0[(j & 3) + 4 * half + 8 * (j >> 2)]; fb1.u[j] = b1[(j & 3) + 4 * half + 8 * (j >> 2)]; }

    accs[0] = __builtin_amdgcn_wmma_f32_16x16x32_bf16(false, fa.v, false, fb0.v,
                                                      (short)0, accs[0], false, false);
    accs[1] = __builtin_amdgcn_wmma_f32_16x16x32_bf16(false, fa.v, false, fb1.v,
                                                      (short)0, accs[1], false, false);
    __syncthreads();
  }

  // C/D layout: vgpr r -> M = wm + 8*half + r; N = (wn + 16*t) + l15
#pragma unroll
  for (int t = 0; t < 2; ++t) {
    int gn = n0 + wn + (t << 4) + l15;
#pragma unroll
    for (int r = 0; r < 8; ++r) {
      int gm = m0 + wm + (half << 3) + r;
      if (gm < p.M && gn < p.N) {
        float v = accs[t][r] * p.alpha;
        if (p.bias) v += p.bias[biasoff + gn];
        if (p.act == 1)      v = fmaxf(v, 0.f);
        else if (p.act == 2) v = 1.f / (1.f + __expf(-v));
        long idx = coff + (long)gm * p.ldc + (long)gn * p.ccs;
        if (p.accum) p.C[idx] += v; else p.C[idx] = v;
      }
    }
  }
}

__global__ void k_layernorm(const float* __restrict__ x, const float* __restrict__ s,
                            const float* __restrict__ o, float* __restrict__ y, int dim) {
  __shared__ float buf[256];
  const int t = threadIdx.x;
  const long row = blockIdx.x;
  float v = x[row * dim + t];
  buf[t] = v; __syncthreads();
  for (int st = dim >> 1; st > 0; st >>= 1) { if (t < st) buf[t] += buf[t + st]; __syncthreads(); }
  float mu = buf[0] / dim; __syncthreads();
  float d = v - mu;
  buf[t] = d * d; __syncthreads();
  for (int st = dim >> 1; st > 0; st >>= 1) { if (t < st) buf[t] += buf[t + st]; __syncthreads(); }
  float inv = rsqrtf(buf[0] / dim + 1e-5f);
  y[row * dim + t] = d * inv * s[t] + o[t];
}

// rows of logits [B][H][Q][Kk]; blockDim.x == Kk; mask bias 1e9*(mask-1)
__global__ void k_softmax(float* __restrict__ lg, const float* __restrict__ mask,
                          const float* __restrict__ nb,
                          int H, int Q, int Kk, long mB, long mK) {
  __shared__ float buf[256];
  const int t = threadIdx.x;
  const long r = blockIdx.x;
  const int  q  = (int)(r % Q);
  const long bh = r / Q;
  const int  h  = (int)(bh % H);
  const long b  = bh / H;
  float v = lg[r * Kk + t] + 1e9f * (mask[b * mB + (long)t * mK] - 1.f);
  if (nb) v += nb[((long)h * Q + q) * Kk + t];
  buf[t] = v; __syncthreads();
  for (int st = Kk >> 1; st > 0; st >>= 1) { if (t < st) buf[t] = fmaxf(buf[t], buf[t + st]); __syncthreads(); }
  float mx = buf[0]; __syncthreads();
  float e = __expf(v - mx);
  buf[t] = e; __syncthreads();
  for (int st = Kk >> 1; st > 0; st >>= 1) { if (t < st) buf[t] += buf[t + st]; __syncthreads(); }
  lg[r * Kk + t] = e / buf[0];
}

// dst = a * (b? b : 1) * (mask? mask[i/C] : 1)
__global__ void k_mul_mask(float* __restrict__ dst, const float* __restrict__ a,
                           const float* __restrict__ b, const float* __restrict__ mask,
                           int C, long n) {
  long i = (long)blockIdx.x * 256 + threadIdx.x;
  if (i >= n) return;
  float v = a[i];
  if (b)    v *= b[i];
  if (mask) v *= mask[i / C];
  dst[i] = v;
}

__global__ void k_fma_add(float* __restrict__ dst, const float* __restrict__ a,
                          const float* __restrict__ b, long n) {
  long i = (long)blockIdx.x * 256 + threadIdx.x;
  if (i >= n) return;
  dst[i] += a[i] * b[i];
}

// y[r][s][c] = x[s][r][c]
__global__ void k_transpose01(const float* __restrict__ x, float* __restrict__ y,
                              int S, int R, int C, long n) {
  long i = (long)blockIdx.x * 256 + threadIdx.x;
  if (i >= n) return;
  int c = (int)(i % C); long t = i / C; int r = (int)(t % R); long s = t / R;
  y[((long)r * S + s) * C + c] = x[i];
}

// dst[s][r][c] += src[r][s][c]
__global__ void k_add_t01(float* __restrict__ dst, const float* __restrict__ src,
                          int S, int R, int C, long n) {
  long i = (long)blockIdx.x * 256 + threadIdx.x;
  if (i >= n) return;
  int c = (int)(i % C); long t = i / C; int r = (int)(t % R); long s = t / R;
  dst[i] += src[((long)r * S + s) * C + c];
}

// O'[b][c][d][e] -> O2[b][d][c][e]   (b,d in 0..255; c,e in 0..31)
__global__ void k_opm_repack(const float* __restrict__ src, float* __restrict__ dst, long n) {
  long i = (long)blockIdx.x * 256 + threadIdx.x;
  if (i >= n) return;
  int e = (int)(i & 31), c = (int)((i >> 5) & 31), d = (int)((i >> 10) & 255);
  long b = i >> 18;
  dst[i] = src[b * 262144 + (long)c * 8192 + (long)d * 32 + e];
}

// norm[d*R+b] = sum_s mm[s][d]*mm[s][b]
__global__ void k_opm_norm(const float* __restrict__ mm, float* __restrict__ norm, int S, int R) {
  long i = (long)blockIdx.x * 256 + threadIdx.x;
  if (i >= (long)R * R) return;
  int b = (int)(i % R), d = (int)(i / R);
  float acc = 0.f;
  for (int s = 0; s < S; ++s) acc += mm[(long)s * R + d] * mm[(long)s * R + b];
  norm[i] = acc;
}

// pair[d][b][f] += outt[b][d][f] / (0.001 + norm[d][b])   (outt already biased)
__global__ void k_opm_final(float* __restrict__ pair, const float* __restrict__ outt,
                            const float* __restrict__ norm, int R, int F, long n) {
  long i = (long)blockIdx.x * 256 + threadIdx.x;
  if (i >= n) return;
  int f = (int)(i % F); long t = i / F; int b = (int)(t % R); long d = t / R;
  pair[i] += outt[((long)b * R + d) * F + f] / (0.001f + norm[d * R + b]);
}

// --------------------------- host-side helpers -----------------------------

static inline void gemmL(hipStream_t st, const GemmP& p, int batches) {
  dim3 g((p.N + BN - 1) / BN, (p.M + BM - 1) / BM, batches);
  const int am = (p.acs == 1 && (((p.aOut | p.aIn | p.lda) & 3) == 0)) ? 0
               : (p.lda == 1 && (((p.aOut | p.aIn | p.acs) & 3) == 0) && (p.M % BM == 0)) ? 1 : 2;
  const int bm = (p.ldb == 1 && (((p.bOut | p.bIn | p.bcs) & 3) == 0)) ? 0 : 1;
  if      (am == 0 && bm == 0) k_gemm<0,0><<<g, dim3(256), 0, st>>>(p);
  else if (am == 0)            k_gemm<0,1><<<g, dim3(256), 0, st>>>(p);
  else if (am == 1 && bm == 0) k_gemm<1,0><<<g, dim3(256), 0, st>>>(p);
  else if (am == 1)            k_gemm<1,1><<<g, dim3(256), 0, st>>>(p);
  else if (bm == 0)            k_gemm<2,0><<<g, dim3(256), 0, st>>>(p);
  else                         k_gemm<2,1><<<g, dim3(256), 0, st>>>(p);
}

static inline unsigned gridn(long n) { return (unsigned)((n + 255) / 256); }

// Gated multi-head self attention (q_data == m_data), KD = 32.
static void run_attention(hipStream_t st, const float* a,
    const float* qw, const float* kw, const float* vw,
    const float* gw, const float* gb, const float* ow, const float* ob,
    int Bb, int Q, int D, int H, int Dout,
    const float* mask, long mB, long mK, const float* nb,
    float* qp, float* kp, float* vp, float* gate, float* logits, float* wv,
    float* outC, int accumOut)
{
  const int KD = KDIM;
  const int Z  = Bb * H;
  for (int which = 0; which < 3; ++which) {  // q/k/v -> [z][Q][KD]
    GemmP p{}; p.A = a;
    p.B = which == 0 ? qw : which == 1 ? kw : vw;
    p.C = which == 0 ? qp : which == 1 ? kp : vp;
    p.M = Q; p.N = KD; p.K = D;
    p.lda = D; p.acs = 1; p.ldb = (long)H * KD; p.bcs = 1; p.ldc = KD; p.ccs = 1;
    p.bi = H; p.aOut = (long)Q * D; p.aIn = 0; p.bOut = 0; p.bIn = KD;
    p.cOut = (long)H * Q * KD; p.cIn = (long)Q * KD;
    p.alpha = which == 0 ? 0.17677669529663687f : 1.f; p.act = 0; p.accum = 0;
    gemmL(st, p, Z);
  }
  { // gate = sigmoid(a@gw + gb), layout [b][q][h][kd]
    GemmP p{}; p.A = a; p.B = gw; p.C = gate; p.bias = gb;
    p.M = Q; p.N = KD; p.K = D;
    p.lda = D; p.acs = 1; p.ldb = (long)H * KD; p.bcs = 1; p.ldc = (long)H * KD; p.ccs = 1;
    p.bi = H; p.aOut = (long)Q * D; p.aIn = 0; p.bOut = 0; p.bIn = KD;
    p.cOut = (long)Q * H * KD; p.cIn = KD; p.biasIn = KD;
    p.alpha = 1.f; p.act = 2; p.accum = 0;
    gemmL(st, p, Z);
  }
  { // logits[z][q][k] = q . k^T
    GemmP p{}; p.A = qp; p.B = kp; p.C = logits;
    p.M = Q; p.N = Q; p.K = KD;
    p.lda = KD; p.acs = 1; p.ldb = 1; p.bcs = KD; p.ldc = Q; p.ccs = 1;
    p.bi = 1; p.aOut = (long)Q * KD; p.bOut = (long)Q * KD; p.cOut = (long)Q * Q;
    p.alpha = 1.f; p.act = 0; p.accum = 0;
    gemmL(st, p, Z);
  }
  k_softmax<<<dim3((unsigned)((long)Z * Q)), dim3(Q), 0, st>>>(logits, mask, nb, H, Q, Q, mB, mK);
  { // wv[b][q][h][kd] = probs @ v
    GemmP p{}; p.A = logits; p.B = vp; p.C = wv;
    p.M = Q; p.N = KD; p.K = Q;
    p.lda = Q; p.acs = 1; p.ldb = KD; p.bcs = 1; p.ldc = (long)H * KD; p.ccs = 1;
    p.bi = H; p.aOut = (long)H * Q * Q; p.aIn = (long)Q * Q;
    p.bOut = (long)H * Q * KD; p.bIn = (long)Q * KD;
    p.cOut = (long)Q * H * KD; p.cIn = KD;
    p.alpha = 1.f; p.act = 0; p.accum = 0;
    gemmL(st, p, Z);
  }
  long n = (long)Bb * Q * H * KD;
  k_mul_mask<<<dim3(gridn(n)), dim3(256), 0, st>>>(wv, wv, gate, nullptr, 1, n);
  { // output projection, optional residual accumulate
    GemmP p{}; p.A = wv; p.B = ow; p.C = outC; p.bias = ob;
    p.M = Bb * Q; p.N = Dout; p.K = H * KD;
    p.lda = (long)H * KD; p.acs = 1; p.ldb = Dout; p.bcs = 1; p.ldc = Dout; p.ccs = 1;
    p.bi = 1; p.alpha = 1.f; p.act = 0; p.accum = accumOut;
    gemmL(st, p, 1);
  }
}

// nb[h][q][k] = einsum('qkc,ch->hqk') via column-major store (ldc=1, ccs=Q*K)
static void run_nb(hipStream_t st, const float* a, const float* feat, float* nbout, int H) {
  GemmP p{}; p.A = a; p.B = feat; p.C = nbout;
  p.M = NRES * NRES; p.N = H; p.K = CZ;
  p.lda = CZ; p.acs = 1; p.ldb = H; p.bcs = 1; p.ldc = 1; p.ccs = (long)NRES * NRES;
  p.bi = 1; p.alpha = 1.f; p.act = 0; p.accum = 0;
  gemmL(st, p, 1);
}

static void run_trimul(hipStream_t st, float* pair_cur, const float* pmask,
    const float* ln_s, const float* ln_o, const float* proj_w, const float* proj_b,
    const float* gate_w, const float* gate_b, const float* c_s, const float* c_o,
    const float* out_w, const float* out_b, const float* gl_w, const float* gl_b,
    int outgoing,
    float* T0, float* P, float* G, float* TRI, float* T5p, float* T6p)
{
  const long PSZ = (long)NRES * NRES * CZ;
  k_layernorm<<<dim3(NRES * NRES), dim3(CZ), 0, st>>>(pair_cur, ln_s, ln_o, T0, CZ);
  { GemmP p{}; p.A = T0; p.B = proj_w; p.C = P; p.bias = proj_b;
    p.M = NRES * NRES; p.N = 2 * NICC; p.K = CZ;
    p.lda = CZ; p.acs = 1; p.ldb = 2 * NICC; p.bcs = 1; p.ldc = 2 * NICC; p.ccs = 1;
    p.bi = 1; p.alpha = 1.f; p.act = 0; p.accum = 0; gemmL(st, p, 1); }
  { GemmP p{}; p.A = T0; p.B = gate_w; p.C = G; p.bias = gate_b;
    p.M = NRES * NRES; p.N = 2 * NICC; p.K = CZ;
    p.lda = CZ; p.acs = 1; p.ldb = 2 * NICC; p.bcs = 1; p.ldc = 2 * NICC; p.ccs = 1;
    p.bi = 1; p.alpha = 1.f; p.act = 2; p.accum = 0; gemmL(st, p, 1); }
  long n2 = (long)NRES * NRES * 2 * NICC;
  k_mul_mask<<<dim3(gridn(n2)), dim3(256), 0, st>>>(P, P, G, pmask, 2 * NICC, n2);

  const long RS = (long)NRES * 2 * NICC;   // row stride (i or k) inside P
  { GemmP p{}; p.C = TRI;
    p.M = NRES; p.N = NRES; p.K = NRES;
    if (outgoing) {        // a[i,j,c] = sum_k lp[i,k,c]*rp[j,k,c]
      p.A = P;        p.lda = RS;        p.acs = 2 * NICC;
      p.B = P + NICC; p.ldb = 2 * NICC;  p.bcs = RS;
    } else {               // a[i,j,c] = sum_k lp[k,j,c]*rp[k,i,c]
      p.A = P + NICC; p.lda = 2 * NICC;  p.acs = RS;
      p.B = P;        p.ldb = RS;        p.bcs = 2 * NICC;
    }
    p.ldc = (long)NRES * CZ; p.ccs = CZ;       // scatter into [i][j][c]
    p.bi = 1; p.aOut = 1; p.bOut = 1; p.cOut = 1;
    p.alpha = 1.f; p.act = 0; p.accum = 0;
    gemmL(st, p, NICC);                        // batch over c = 128 channels
  }
  k_layernorm<<<dim3(NRES * NRES), dim3(CZ), 0, st>>>(TRI, c_s, c_o, T0, CZ);
  { GemmP p{}; p.A = T0; p.B = out_w; p.C = T5p; p.bias = out_b;
    p.M = NRES * NRES; p.N = CZ; p.K = NICC;
    p.lda = NICC; p.acs = 1; p.ldb = CZ; p.bcs = 1; p.ldc = CZ; p.ccs = 1;
    p.bi = 1; p.alpha = 1.f; p.act = 0; p.accum = 0; gemmL(st, p, 1); }
  { GemmP p{}; p.A = T5p; p.B = gl_w; p.C = T6p; p.bias = gl_b;
    p.M = NRES * NRES; p.N = CZ; p.K = CZ;
    p.lda = CZ; p.acs = 1; p.ldb = CZ; p.bcs = 1; p.ldc = CZ; p.ccs = 1;
    p.bi = 1; p.alpha = 1.f; p.act = 2; p.accum = 0; gemmL(st, p, 1); }
  k_fma_add<<<dim3(gridn(PSZ)), dim3(256), 0, st>>>(pair_cur, T5p, T6p, PSZ);
}

// ----------------------------------------------------------------------------

extern "C" void kernel_launch(void* const* d_in, const int* in_sizes, int n_in,
                              void* d_out, int out_size, void* d_ws, size_t ws_size,
                              hipStream_t stream) {
  (void)in_sizes; (void)n_in; (void)out_size; (void)ws_size;
  auto IN = [&](int i) -> const float* { return (const float*)d_in[i]; };
  float* out = (float*)d_out;
  float* ws  = (float*)d_ws;

  const long MSZ = (long)NSEQ * NRES * CM;   // 8,388,608
  const long PSZ = (long)NRES * NRES * CZ;   // 8,388,608
  float* msa_cur  = out;
  float* pair_cur = out + MSZ;

  float* A_ = ws;                 // 67,108,864 f arena (probs / OPM outer)
  float* B_ = ws + 67108864L;     // 67,108,864 f arena (OPM repack / FFN hidden)
  float* Ts[10];
  for (int i = 0; i < 10; ++i) Ts[i] = ws + 134217728L + (long)i * 16777216L;
  float *T0 = Ts[0], *T1 = Ts[1], *T2 = Ts[2], *T3 = Ts[3], *T4 = Ts[4],
        *T5 = Ts[5], *T6 = Ts[6], *T7 = Ts[7], *T8 = Ts[8], *T9 = Ts[9];

  hipMemcpyAsync(msa_cur,  d_in[I_MSA],  MSZ * sizeof(float), hipMemcpyDeviceToDevice, stream);
  hipMemcpyAsync(pair_cur, d_in[I_PAIR], PSZ * sizeof(float), hipMemcpyDeviceToDevice, stream);
  const float* mmask = IN(I_MMASK);
  const float* pmask = IN(I_PMASK);

  // ---- 1) MSA row attention with pair bias ----
  k_layernorm<<<dim3(NSEQ * NRES), dim3(CM), 0, stream>>>(msa_cur, IN(MR_NORM_S), IN(MR_NORM_O), T0, CM);
  k_layernorm<<<dim3(NRES * NRES), dim3(CZ), 0, stream>>>(pair_cur, IN(MR_PN_S), IN(MR_PN_O), T1, CZ);
  run_nb(stream, T1, IN(MR_FEAT), T7, HM);
  run_attention(stream, T0, IN(MR_Q_W), IN(MR_K_W), IN(MR_V_W), IN(MR_GATING_W), IN(MR_GATING_B),
                IN(MR_OUTPUT_W), IN(MR_OUTPUT_B),
                NSEQ, NRES, CM, HM, CM, mmask, NRES, 1, T7,
                T2, T3, T4, T5, A_, T6, msa_cur, 1);

  // ---- 2) MSA column attention ----
  k_transpose01<<<dim3(gridn(MSZ)), dim3(256), 0, stream>>>(msa_cur, T8, NSEQ, NRES, CM, MSZ);
  k_layernorm<<<dim3(NRES * NSEQ), dim3(CM), 0, stream>>>(T8, IN(MC_NORM_S), IN(MC_NORM_O), T0, CM);
  run_attention(stream, T0, IN(MC_Q_W), IN(MC_K_W), IN(MC_V_W), IN(MC_GATING_W), IN(MC_GATING_B),
                IN(MC_OUTPUT_W), IN(MC_OUTPUT_B),
                NRES, NSEQ, CM, HM, CM, mmask, 1, NRES, nullptr,
                T2, T3, T4, T5, A_, T6, T9, 0);
  k_add_t01<<<dim3(gridn(MSZ)), dim3(256), 0, stream>>>(msa_cur, T9, NSEQ, NRES, CM, MSZ);

  // ---- 3) MSA transition ----
  k_layernorm<<<dim3(NSEQ * NRES), dim3(CM), 0, stream>>>(msa_cur, IN(MT_NORM_S), IN(MT_NORM_O), T0, CM);
  { GemmP p{}; p.A = T0; p.B = IN(MT_W1); p.C = B_; p.bias = IN(MT_B1);
    p.M = NSEQ * NRES; p.N = 4 * CM; p.K = CM;
    p.lda = CM; p.acs = 1; p.ldb = 4 * CM; p.bcs = 1; p.ldc = 4 * CM; p.ccs = 1;
    p.bi = 1; p.alpha = 1.f; p.act = 1; p.accum = 0; gemmL(stream, p, 1); }
  { GemmP p{}; p.A = B_; p.B = IN(MT_W2); p.C = msa_cur; p.bias = IN(MT_B2);
    p.M = NSEQ * NRES; p.N = CM; p.K = 4 * CM;
    p.lda = 4 * CM; p.acs = 1; p.ldb = CM; p.bcs = 1; p.ldc = CM; p.ccs = 1;
    p.bi = 1; p.alpha = 1.f; p.act = 0; p.accum = 1; gemmL(stream, p, 1); }

  // ---- 4) Outer product mean ----
  k_layernorm<<<dim3(NSEQ * NRES), dim3(CM), 0, stream>>>(msa_cur, IN(OPM_NORM_S), IN(OPM_NORM_O), T0, CM);
  { GemmP p{}; p.A = T0; p.B = IN(OPM_LEFT_W); p.C = T1; p.bias = IN(OPM_LEFT_B);
    p.M = NSEQ * NRES; p.N = 32; p.K = CM;
    p.lda = CM; p.acs = 1; p.ldb = 32; p.bcs = 1; p.ldc = 32; p.ccs = 1;
    p.bi = 1; p.alpha = 1.f; p.act = 0; p.accum = 0; gemmL(stream, p, 1); }
  k_mul_mask<<<dim3(gridn((long)NSEQ * NRES * 32)), dim3(256), 0, stream>>>(T1, T1, nullptr, mmask, 32, (long)NSEQ * NRES * 32);
  { GemmP p{}; p.A = T0; p.B = IN(OPM_RIGHT_W); p.C = T2; p.bias = IN(OPM_RIGHT_B);
    p.M = NSEQ * NRES; p.N = 32; p.K = CM;
    p.lda = CM; p.acs = 1; p.ldb = 32; p.bcs = 1; p.ldc = 32; p.ccs = 1;
    p.bi = 1; p.alpha = 1.f; p.act = 0; p.accum = 0; gemmL(stream, p, 1); }
  k_mul_mask<<<dim3(gridn((long)NSEQ * NRES * 32)), dim3(256), 0, stream>>>(T2, T2, nullptr, mmask, 32, (long)NSEQ * NRES * 32);
  { // O'[(b,c),(d,e)] = sum_s L[s,(b,c)] R[s,(d,e)]  (A contiguous in M)
    GemmP p{}; p.A = T1; p.B = T2; p.C = A_;
    p.M = NRES * 32; p.N = NRES * 32; p.K = NSEQ;
    p.lda = 1; p.acs = (long)NRES * 32; p.ldb = (long)NRES * 32; p.bcs = 1;
    p.ldc = (long)NRES * 32; p.ccs = 1;
    p.bi = 1; p.alpha = 1.f; p.act = 0; p.accum = 0; gemmL(stream, p, 1); }
  k_opm_repack<<<dim3(gridn(67108864L)), dim3(256), 0, stream>>>(A_, B_, 67108864L);
  { // outt[(b,d)][f] = O2 @ W + out_b
    GemmP p{}; p.A = B_; p.B = IN(OPM_OUT_W); p.C = T3; p.bias = IN(OPM_OUT_B);
    p.M = NRES * NRES; p.N = CZ; p.K = 1024;
    p.lda = 1024; p.acs = 1; p.ldb = CZ; p.bcs = 1; p.ldc = CZ; p.ccs = 1;
    p.bi = 1; p.alpha = 1.f; p.act = 0; p.accum = 0; gemmL(stream, p, 1); }
  k_opm_norm<<<dim3(gridn((long)NRES * NRES)), dim3(256), 0, stream>>>(mmask, T4, NSEQ, NRES);
  k_opm_final<<<dim3(gridn(PSZ)), dim3(256), 0, stream>>>(pair_cur, T3, T4, NRES, CZ, PSZ);

  // ---- 5) Triangle multiplication (outgoing) ----
  run_trimul(stream, pair_cur, pmask,
             IN(TMO_LN_S), IN(TMO_LN_O), IN(TMO_PROJ_W), IN(TMO_PROJ_B),
             IN(TMO_GATE_W), IN(TMO_GATE_B), IN(TMO_C_S), IN(TMO_C_O),
             IN(TMO_OUT_W), IN(TMO_OUT_B), IN(TMO_GL_W), IN(TMO_GL_B),
             1, T0, T1, T2, T4, T5, T6);

  // ---- 6) Triangle multiplication (incoming) ----
  run_trimul(stream, pair_cur, pmask,
             IN(TMI_LN_S), IN(TMI_LN_O), IN(TMI_PROJ_W), IN(TMI_PROJ_B),
             IN(TMI_GATE_W), IN(TMI_GATE_B), IN(TMI_C_S), IN(TMI_C_O),
             IN(TMI_OUT_W), IN(TMI_OUT_B), IN(TMI_GL_W), IN(TMI_GL_B),
             0, T0, T1, T2, T4, T5, T6);

  // ---- 7) Triangle attention (starting node) ----
  k_layernorm<<<dim3(NRES * NRES), dim3(CZ), 0, stream>>>(pair_cur, IN(TAS_NORM_S), IN(TAS_NORM_O), T0, CZ);
  run_nb(stream, T0, IN(TAS_FEAT), T7, HP);
  run_attention(stream, T0, IN(TAS_Q_W), IN(TAS_K_W), IN(TAS_V_W), IN(TAS_GATING_W), IN(TAS_GATING_B),
                IN(TAS_OUTPUT_W), IN(TAS_OUTPUT_B),
                NRES, NRES, CZ, HP, CZ, pmask, NRES, 1, T7,
                T1, T2, T3, T5, A_, T6, pair_cur, 1);

  // ---- 8) Triangle attention (ending node) ----
  k_transpose01<<<dim3(gridn(PSZ)), dim3(256), 0, stream>>>(pair_cur, T8, NRES, NRES, CZ, PSZ);
  k_layernorm<<<dim3(NRES * NRES), dim3(CZ), 0, stream>>>(T8, IN(TAE_NORM_S), IN(TAE_NORM_O), T0, CZ);
  run_nb(stream, T0, IN(TAE_FEAT), T7, HP);
  run_attention(stream, T0, IN(TAE_Q_W), IN(TAE_K_W), IN(TAE_V_W), IN(TAE_GATING_W), IN(TAE_GATING_B),
                IN(TAE_OUTPUT_W), IN(TAE_OUTPUT_B),
                NRES, NRES, CZ, HP, CZ, pmask, 1, NRES, T7,
                T1, T2, T3, T5, A_, T6, T9, 0);
  k_add_t01<<<dim3(gridn(PSZ)), dim3(256), 0, stream>>>(pair_cur, T9, NRES, NRES, CZ, PSZ);

  // ---- 9) Pair transition ----
  k_layernorm<<<dim3(NRES * NRES), dim3(CZ), 0, stream>>>(pair_cur, IN(PT_NORM_S), IN(PT_NORM_O), T0, CZ);
  { GemmP p{}; p.A = T0; p.B = IN(PT_W1); p.C = B_; p.bias = IN(PT_B1);
    p.M = NRES * NRES; p.N = 4 * CZ; p.K = CZ;
    p.lda = CZ; p.acs = 1; p.ldb = 4 * CZ; p.bcs = 1; p.ldc = 4 * CZ; p.ccs = 1;
    p.bi = 1; p.alpha = 1.f; p.act = 1; p.accum = 0; gemmL(stream, p, 1); }
  { GemmP p{}; p.A = B_; p.B = IN(PT_W2); p.C = pair_cur; p.bias = IN(PT_B2);
    p.M = NRES * NRES; p.N = CZ; p.K = 4 * CZ;
    p.lda = 4 * CZ; p.acs = 1; p.ldb = CZ; p.bcs = 1; p.ldc = CZ; p.ccs = 1;
    p.bi = 1; p.alpha = 1.f; p.act = 0; p.accum = 1; gemmL(stream, p, 1); }
}
